// LambdaNetwork1DBase_36429912604939
// MI455X (gfx1250) — matrix operand order repaired and checked by reading, not compiled, not run on hardware
//
#include <hip/hip_runtime.h>
#include <hip/hip_bf16.h>
#include <stdint.h>

// ---------------- problem constants ----------------
constexpr int BB     = 16;
constexpr int TT     = 1024;
constexpr int DIN    = 1024;
constexpr int HEADS  = 8;
constexpr int KD     = 64;
constexpr int VD     = 128;
constexpr int M      = BB * TT;        // 16384 rows
constexpr int NQ     = HEADS * KD;     // 512 q channels
constexpr int NV     = VD;             // 128 v channels
constexpr int NN     = NQ + NV;        // 640 fused output channels
constexpr int KK     = DIN;            // 1024 reduction
constexpr int KSTEP  = 32;             // bf16 WMMA K
constexpr int KSTEPS = KK / KSTEP;     // 32
constexpr int MT     = M / 16;         // 1024 M tiles
constexpr int NT     = NN / 16;        // 40 N tiles
constexpr float EPS  = 1e-3f;

// ---------------- WMMA types ----------------
typedef __attribute__((ext_vector_type(16))) __bf16 v16bf;
typedef __attribute__((ext_vector_type(8)))  float  v8f;

union Frag { v16bf v; uint4 q[2]; };

// ---------------- helpers ----------------
static __device__ __forceinline__ uint16_t f32_to_bf16(float f) {
    uint32_t u = __float_as_uint(f);
    uint32_t r = u + 0x7fffu + ((u >> 16) & 1u);   // round-to-nearest-even
    return (uint16_t)(r >> 16);
}
static __device__ __forceinline__ float bf16_to_f32(uint16_t h) {
    return __uint_as_float(((uint32_t)h) << 16);
}

// =======================================================================
// K0: pack x -> A fragments (bf16 hi/lo) in CDNA5 16-bit A-matrix layout.
// A tile 16x32: lane l holds row M=l&15; h=l>>4; VGPR j pair covers
// K = ((j>>2)<<4) + ((j&3)<<1) + (h<<3)  (ISA 7.12.2 table).
// =======================================================================
__global__ __launch_bounds__(256) void lam_prep_x(
    const float* __restrict__ x, uint32_t* __restrict__ Ahi, uint32_t* __restrict__ Alo)
{
    int tid  = blockIdx.x * 256 + threadIdx.x;        // MT*KSTEPS*32 = 1,048,576
    int lane = tid & 31;
    int ks   = (tid >> 5) & (KSTEPS - 1);
    int mt   = tid >> 10;
    if (mt >= MT) return;
    int row = mt * 16 + (lane & 15);
    int h   = lane >> 4;
    const float* xr = x + (size_t)row * KK + ks * KSTEP;
    uint32_t hi[8], lo[8];
#pragma unroll
    for (int j = 0; j < 8; ++j) {
        int k0 = ((j >> 2) << 4) + ((j & 3) << 1) + (h << 3);
        float f0 = xr[k0], f1 = xr[k0 + 1];
        uint16_t h0 = f32_to_bf16(f0), h1 = f32_to_bf16(f1);
        uint16_t l0 = f32_to_bf16(f0 - bf16_to_f32(h0));
        uint16_t l1 = f32_to_bf16(f1 - bf16_to_f32(h1));
        hi[j] = (uint32_t)h0 | ((uint32_t)h1 << 16);
        lo[j] = (uint32_t)l0 | ((uint32_t)l1 << 16);
    }
    size_t o = ((size_t)(mt * KSTEPS + ks) * 32 + lane) * 8;
    uint4* ph = (uint4*)(Ahi + o);
    uint4* pl = (uint4*)(Alo + o);
    ph[0] = make_uint4(hi[0], hi[1], hi[2], hi[3]);
    ph[1] = make_uint4(hi[4], hi[5], hi[6], hi[7]);
    pl[0] = make_uint4(lo[0], lo[1], lo[2], lo[3]);
    pl[1] = make_uint4(lo[4], lo[5], lo[6], lo[7]);
}

// =======================================================================
// K1: pack [Wq | Wv] -> B fragments (bf16 hi/lo), B-matrix 32x16 layout:
// lane l holds column N=l&15; h=l>>4; VGPR j pair covers
// K = ((j>>2)<<4) + ((j&3)<<1) + (h<<3)  (same as A).
// =======================================================================
__global__ __launch_bounds__(256) void lam_prep_w(
    const float* __restrict__ Wq, const float* __restrict__ Wv,
    uint32_t* __restrict__ Bhi, uint32_t* __restrict__ Blo)
{
    int tid  = blockIdx.x * 256 + threadIdx.x;        // NT*KSTEPS*32 = 40,960
    int lane = tid & 31;
    int ks   = (tid >> 5) & (KSTEPS - 1);
    int nt   = tid >> 10;
    if (nt >= NT) return;
    int n = nt * 16 + (lane & 15);
    int h = lane >> 4;
    uint32_t hi[8], lo[8];
#pragma unroll
    for (int j = 0; j < 8; ++j) {
        uint32_t hw[2], lw[2];
#pragma unroll
        for (int e = 0; e < 2; ++e) {
            int kk = ks * KSTEP + ((j >> 2) << 4) + ((j & 3) << 1) + (h << 3) + e;
            float f = (n < NQ) ? Wq[(size_t)kk * NQ + n]
                               : Wv[(size_t)kk * NV + (n - NQ)];
            uint16_t fh = f32_to_bf16(f);
            uint16_t fl = f32_to_bf16(f - bf16_to_f32(fh));
            hw[e] = fh; lw[e] = fl;
        }
        hi[j] = hw[0] | (hw[1] << 16);
        lo[j] = lw[0] | (lw[1] << 16);
    }
    size_t o = ((size_t)(nt * KSTEPS + ks) * 32 + lane) * 8;
    uint4* ph = (uint4*)(Bhi + o);
    uint4* pl = (uint4*)(Blo + o);
    ph[0] = make_uint4(hi[0], hi[1], hi[2], hi[3]);
    ph[1] = make_uint4(hi[4], hi[5], hi[6], hi[7]);
    pl[0] = make_uint4(lo[0], lo[1], lo[2], lo[3]);
    pl[1] = make_uint4(lo[4], lo[5], lo[6], lo[7]);
}

// =======================================================================
// K2: w[t] = 1 + sum_k E[t,t,k]   (E diag is 64 contiguous floats)
// =======================================================================
__global__ __launch_bounds__(256) void lam_prep_e(
    const float* __restrict__ E, float* __restrict__ wt)
{
    int t = blockIdx.x * 256 + threadIdx.x;
    if (t >= TT) return;
    const float4* p = (const float4*)(E + (size_t)t * (TT + 1) * KD);
    float acc = 0.f;
#pragma unroll
    for (int i = 0; i < KD / 4; ++i) {
        float4 f = p[i];
        acc += f.x + f.y + f.z + f.w;
    }
    wt[t] = 1.f + acc;
}

// =======================================================================
// K3: fused GEMM  qv[16384,640] = x @ [Wq|Wv]  via split-bf16 WMMA,
// with per-channel sum/sumsq accumulation for BatchNorm stats.
// Block = 256 threads = 8 waves (4 M x 2 N); wave = 32x32 C (2x2 tiles).
// Block tile = 128(M) x 64(N); grid = (128, 10).
// =======================================================================
__global__ __launch_bounds__(256) void lam_gemm(
    const uint4* __restrict__ Ahi, const uint4* __restrict__ Alo,
    const uint4* __restrict__ Bhi, const uint4* __restrict__ Blo,
    float* __restrict__ qv, float* __restrict__ colsum, float* __restrict__ colsq)
{
    __shared__ float lsum[64];
    __shared__ float lsq[64];
    const int tid = threadIdx.x;
    if (tid < 64) { lsum[tid] = 0.f; lsq[tid] = 0.f; }
    __syncthreads();

    const int lane = tid & 31;
    const int w    = tid >> 5;
    const int wm   = w & 3;          // 0..3
    const int wn   = w >> 2;         // 0..1
    const int mt0  = blockIdx.x * 8 + wm * 2;   // M tile index
    const int nt0  = blockIdx.y * 4 + wn * 2;   // N tile index

    v8f c[2][2] = {};

    for (int ks = 0; ks < KSTEPS; ++ks) {
        Frag ah[2], al[2], bh[2], bl[2];
#pragma unroll
        for (int i = 0; i < 2; ++i) {
            size_t ai = ((size_t)((mt0 + i) * KSTEPS + ks) * 32 + lane) * 2;
            ah[i].q[0] = Ahi[ai]; ah[i].q[1] = Ahi[ai + 1];
            al[i].q[0] = Alo[ai]; al[i].q[1] = Alo[ai + 1];
            size_t bi = ((size_t)((nt0 + i) * KSTEPS + ks) * 32 + lane) * 2;
            bh[i].q[0] = Bhi[bi]; bh[i].q[1] = Bhi[bi + 1];
            bl[i].q[0] = Blo[bi]; bl[i].q[1] = Blo[bi + 1];
        }
#pragma unroll
        for (int i = 0; i < 2; ++i) {
#pragma unroll
            for (int j = 0; j < 2; ++j) {
                c[i][j] = __builtin_amdgcn_wmma_f32_16x16x32_bf16(
                    false, ah[i].v, false, bh[j].v, (short)0, c[i][j], false, false);
                c[i][j] = __builtin_amdgcn_wmma_f32_16x16x32_bf16(
                    false, al[i].v, false, bh[j].v, (short)0, c[i][j], false, false);
                c[i][j] = __builtin_amdgcn_wmma_f32_16x16x32_bf16(
                    false, ah[i].v, false, bl[j].v, (short)0, c[i][j], false, false);
            }
        }
    }

    // C layout: VGPR r -> lanes 0-15: (M=r, N=lane); lanes 16-31: (M=r+8, N=lane-16)
    const int rbase = (lane >> 4) << 3;
    const int cl    = lane & 15;
#pragma unroll
    for (int i = 0; i < 2; ++i) {
#pragma unroll
        for (int j = 0; j < 2; ++j) {
            int col = (nt0 + j) * 16 + cl;
            float ps = 0.f, pq = 0.f;
#pragma unroll
            for (int r = 0; r < 8; ++r) {
                float val = c[i][j][r];
                int row = (mt0 + i) * 16 + rbase + r;
                qv[(size_t)row * NN + col] = val;
                ps += val;
                pq += val * val;
            }
            int lcol = (wn * 2 + j) * 16 + cl;
            atomicAdd(&lsum[lcol], ps);
            atomicAdd(&lsq[lcol], pq);
        }
    }
    __syncthreads();
    if (tid < 64) {
        atomicAdd(&colsum[blockIdx.y * 64 + tid], lsum[tid]);
        atomicAdd(&colsq [blockIdx.y * 64 + tid], lsq [tid]);
    }
}

// =======================================================================
// K4: BN coefficients a_c, b_c; per-head bias sums; Wsum = sum_t w_t.
// =======================================================================
__global__ __launch_bounds__(640) void lam_finalize(
    const float* __restrict__ colsum, const float* __restrict__ colsq,
    const float* __restrict__ gq, const float* __restrict__ bq,
    const float* __restrict__ gv, const float* __restrict__ bv,
    const float* __restrict__ wt,
    float* __restrict__ avec, float* __restrict__ bvec, float* __restrict__ hb)
{
    int c = threadIdx.x;
    float mean = colsum[c] * (1.f / (float)M);
    float var  = colsq[c]  * (1.f / (float)M) - mean * mean;
    float g  = (c < NQ) ? gq[c] : gv[c - NQ];
    float be = (c < NQ) ? bq[c] : bv[c - NQ];
    float a  = g * rsqrtf(var + EPS);
    avec[c] = a;
    bvec[c] = be - a * mean;
    __syncthreads();
    if (c < HEADS) {
        float s = 0.f;
        for (int j = 0; j < KD; ++j) s += bvec[c * KD + j];
        hb[c] = s;
    } else if (c == HEADS) {
        float s = 0.f;
        for (int t = 0; t < TT; ++t) s += wt[t];
        hb[8] = s;   // Wsum
    }
}

// =======================================================================
// K5: q_sum[m,h] = sum_{c in head h} a_c * qv[m,c]  + headbias[h]
// one wave per row; lane reads channels lane+32*i (head = i>>1).
// =======================================================================
__global__ __launch_bounds__(256) void lam_qsum(
    const float* __restrict__ qv, const float* __restrict__ avec,
    const float* __restrict__ hb, float* __restrict__ qsum)
{
    int m    = blockIdx.x * 8 + (threadIdx.x >> 5);
    int lane = threadIdx.x & 31;
    const float* row = qv + (size_t)m * NN;
    float p[8];
#pragma unroll
    for (int h = 0; h < 8; ++h) p[h] = 0.f;
#pragma unroll
    for (int i = 0; i < 16; ++i) {
        int ch = lane + 32 * i;
        p[i >> 1] += avec[ch] * row[ch];
    }
#pragma unroll
    for (int h = 0; h < 8; ++h) {
#pragma unroll
        for (int off = 16; off > 0; off >>= 1)
            p[h] += __shfl_xor(p[h], off, 32);
    }
    float out = p[0];
#pragma unroll
    for (int h = 1; h < 8; ++h) out = (lane == h) ? p[h] : out;
    if (lane < 8) qsum[(size_t)m * 8 + lane] = out + hb[lane];
}

// =======================================================================
// K6: s[b,v] = a_v * sum_t w_t * qv[b,t,NQ+v] + b_v * Wsum
// =======================================================================
__global__ __launch_bounds__(1024) void lam_svec(
    const float* __restrict__ qv, const float* __restrict__ wt,
    const float* __restrict__ avec, const float* __restrict__ bvec,
    const float* __restrict__ hb, float* __restrict__ svec)
{
    __shared__ float red[8][128];
    int b  = blockIdx.x;
    int v  = threadIdx.x & 127;
    int ty = threadIdx.x >> 7;
    float acc = 0.f;
    for (int t = ty; t < TT; t += 8)
        acc += wt[t] * qv[(size_t)(b * TT + t) * NN + NQ + v];
    red[ty][v] = acc;
    __syncthreads();
    if (ty == 0) {
        float tot = acc;
#pragma unroll
        for (int r = 1; r < 8; ++r) tot += red[r][v];
        svec[b * 128 + v] = avec[NQ + v] * tot + bvec[NQ + v] * hb[8];
    }
}

// =======================================================================
// K7: Y[b,t,h,v] = qsum[b,t,h] * s[b,v]   (output [16,1024,1024] f32)
// =======================================================================
__global__ __launch_bounds__(256) void lam_out(
    const float* __restrict__ qsum, const float* __restrict__ svec,
    float* __restrict__ out)
{
    size_t idx = (size_t)blockIdx.x * 256 + threadIdx.x;
    int v  = (int)(idx & 127);
    int h  = (int)((idx >> 7) & 7);
    size_t bt = idx >> 10;
    int b  = (int)(bt >> 10);
    out[idx] = qsum[bt * 8 + h] * svec[b * 128 + v];
}

// =======================================================================
// launch
// =======================================================================
extern "C" void kernel_launch(void* const* d_in, const int* in_sizes, int n_in,
                              void* d_out, int out_size, void* d_ws, size_t ws_size,
                              hipStream_t stream) {
    const float* x  = (const float*)d_in[0];
    const float* Wq = (const float*)d_in[1];
    // d_in[2] = Wk: mathematically irrelevant (softmax rows sum to 1) -> skipped
    const float* Wv = (const float*)d_in[3];
    const float* gq = (const float*)d_in[4];
    const float* bq = (const float*)d_in[5];
    const float* gv = (const float*)d_in[6];
    const float* bv = (const float*)d_in[7];
    const float* E  = (const float*)d_in[8];

    char* ws = (char*)d_ws;
    const size_t sA  = (size_t)M * KK * 2;         // 33.5 MB per A buffer
    const size_t sB  = (size_t)NN * KK * 2;        // 1.31 MB per B buffer
    const size_t sQV = (size_t)M * NN * 4;         // 41.9 MB

    size_t off = 0;
    uint32_t* Ahi = (uint32_t*)(ws + off); off += sA;
    uint32_t* Alo = (uint32_t*)(ws + off); off += sA;
    uint32_t* Bhi = (uint32_t*)(ws + off); off += sB;
    uint32_t* Blo = (uint32_t*)(ws + off); off += sB;
    float* qv     = (float*)(ws + off);    off += sQV;
    float* colsum = (float*)(ws + off);    off += NN * sizeof(float);
    float* colsq  = (float*)(ws + off);    off += NN * sizeof(float);
    float* avec   = (float*)(ws + off);    off += NN * sizeof(float);
    float* bvec   = (float*)(ws + off);    off += NN * sizeof(float);
    float* hb     = (float*)(ws + off);    off += 64 * sizeof(float);
    float* wt     = (float*)(ws + off);    off += TT * sizeof(float);
    float* qsum   = (float*)(ws + off);    off += (size_t)M * 8 * sizeof(float);
    float* svec   = (float*)(ws + off);    off += BB * 128 * sizeof(float);

    // zero BN stat accumulators (colsum & colsq are adjacent)
    (void)hipMemsetAsync(colsum, 0, 2 * NN * sizeof(float), stream);

    lam_prep_x<<<(MT * KSTEPS * 32) / 256, 256, 0, stream>>>(x, Ahi, Alo);
    lam_prep_w<<<(NT * KSTEPS * 32) / 256, 256, 0, stream>>>(Wq, Wv, Bhi, Blo);
    lam_prep_e<<<TT / 256, 256, 0, stream>>>(E, wt);

    lam_gemm<<<dim3(MT / 8, NT / 4), 256, 0, stream>>>(
        (const uint4*)Ahi, (const uint4*)Alo,
        (const uint4*)Bhi, (const uint4*)Blo, qv, colsum, colsq);

    lam_finalize<<<1, NN, 0, stream>>>(colsum, colsq, gq, bq, gv, bv, wt, avec, bvec, hb);
    lam_qsum<<<M / 8, 256, 0, stream>>>(qv, avec, hb, qsum);
    lam_svec<<<BB, 1024, 0, stream>>>(qv, wt, avec, bvec, hb, svec);
    lam_out<<<(int)(((size_t)M * 1024) / 256), 256, 0, stream>>>(qsum, svec, (float*)d_out);
}
